// MoETransformer_39384850104910
// MI455X (gfx1250) — compile-verified, matrix operand order checked by reading, not run on hardware
//
#include <hip/hip_runtime.h>
#include <hip/hip_bf16.h>
#include <math.h>

// ---------------- problem constants ----------------
constexpr int N_TOK  = 4096;   // B*T
constexpr int C_DIM  = 1024;   // d_model
constexpr int E_NUM  = 8;      // experts
constexpr int H_DIM  = 512;    // expert dim
constexpr int HS_DIM = 2048;   // shared expert dim
constexpr int CAP    = 4096;   // worst-case tokens per expert

// ---------------- GEMM tiling ----------------
constexpr int BM  = 64;        // block tile M
constexpr int BN  = 128;       // block tile N
constexpr int BK  = 32;        // K per stage == WMMA K
constexpr int LDT = BK + 8;    // LDS row stride in bf16 (80B: 16B-aligned, bank-spread)

typedef __attribute__((ext_vector_type(16))) __bf16 v16bf;
typedef __attribute__((ext_vector_type(8)))  float  v8f;
typedef __attribute__((ext_vector_type(4)))  unsigned int u32x4;
typedef __attribute__((ext_vector_type(2)))  unsigned int u32x2;
typedef __attribute__((ext_vector_type(4)))  float f32x4;
typedef __attribute__((ext_vector_type(8)))  int   i32x8;
typedef __attribute__((ext_vector_type(4)))  int   i32x4;

// sigmoid via native v_exp_f32 + v_rcp_f32 (avoid IEEE div expansion in epilogue)
__device__ inline float sigmoid_f(float x) {
  return __builtin_amdgcn_rcpf(1.0f + __expf(-x));
}
__device__ inline float silu_f(float x) { return x * sigmoid_f(x); }

// pack two f32 -> packed bf16 pair
__device__ inline unsigned int pk2bf(float a, float b) {
#if __has_builtin(__builtin_amdgcn_cvt_pk_bf16_f32)
  auto r = __builtin_amdgcn_cvt_pk_bf16_f32(a, b);
  unsigned int u;
  __builtin_memcpy(&u, &r, 4);
  return u;
#else
  __bf16 lo = (__bf16)a, hi = (__bf16)b;
  unsigned short l, h;
  __builtin_memcpy(&l, &lo, 2);
  __builtin_memcpy(&h, &hi, 2);
  return (unsigned int)l | ((unsigned int)h << 16);
#endif
}

// Async memory->LDS copy, 16B per lane (GLOBAL_LOAD_ASYNC_TO_LDS_B128, ASYNCcnt).
__device__ inline void async_copy16(void* lds_dst, const void* gsrc) {
  unsigned int       l = (unsigned int)(unsigned long long)(uintptr_t)lds_dst;
  unsigned long long g = (unsigned long long)(uintptr_t)gsrc;
  asm volatile("global_load_async_to_lds_b128 %0, %1, off"
               :: "v"(l), "v"(g) : "memory");
}
__device__ inline void wait_async0() {
  asm volatile("s_wait_asynccnt 0x0" ::: "memory");
}

// ---- Tensor Data Mover: 2D bf16 tile (tile_x elems wide, tile_y rows) from a
// row-major matrix with row stride `row_stride_elems` into LDS, with hardware
// padding of 16B after each 64B row to match the LDT=40 layout.
// D# packing per CDNA5 ISA ch.8 (groups 0/1; groups 2/3 zero for 2D tensors).
// This toolchain declares the 6-arg builtin (extra i32x8 group + cpol).
__device__ inline void tdm_load_tile_2d(const void* gsrc, void* lds_dst,
                                        unsigned int tile_x_elems,
                                        unsigned int tile_y,
                                        unsigned int row_stride_elems) {
  unsigned long long ga  = (unsigned long long)(uintptr_t)gsrc;
  unsigned int       lds = (unsigned int)(uintptr_t)lds_dst;
  u32x4 g0;
  g0.x = 1u;                                                    // count=1 (valid), user
  g0.y = lds;                                                   // lds_addr
  g0.z = (unsigned int)(ga & 0xffffffffull);                    // global_addr[31:0]
  g0.w = (unsigned int)((ga >> 32) & 0x01ffffffull) | (2u << 30); // addr[56:32] | type=2
  unsigned int td0 = row_stride_elems;                          // tensor_dim0
  unsigned int td1 = (unsigned int)N_TOK;                       // tensor_dim1
  i32x8 g1;
  g1[0] = (int)((1u << 16)      // data_size = 2 bytes
              | (1u << 20)      // pad_enable
              | (3u << 22)      // pad_interval: 16 DWORDs (64B) between pads
              | (3u << 25));    // pad_amount : 4 DWORDs (16B)
  g1[1] = (int)((td0 & 0xffffu) << 16);                         // tensor_dim0[15:0]
  g1[2] = (int)((td0 >> 16) | ((td1 & 0xffffu) << 16));         // dim0 hi | dim1 lo
  g1[3] = (int)((td1 >> 16) | (tile_x_elems << 16));            // dim1 hi | tile_dim0
  g1[4] = (int)tile_y;                                          // tile_dim1 (tile_dim2=0)
  g1[5] = (int)row_stride_elems;                                // tensor_dim0_stride lo
  g1[6] = 0;                                                    // stride0 hi | stride1 lo
  g1[7] = 0;                                                    // stride1 hi
  i32x4 z4 = {0, 0, 0, 0};
  i32x8 z8 = {0, 0, 0, 0, 0, 0, 0, 0};
  __builtin_amdgcn_tensor_load_to_lds(g0, g1, z4, z4, z8, 0);
}
__device__ inline void wait_tensor0() {
  __builtin_amdgcn_s_wait_tensorcnt(0);
}

// A/B fragment: lane (0..15) holds row (A: M, B: N) = row0+lane, K=0..15;
// lane+16 holds same row, K=16..31. 16 contiguous bf16 per lane.
__device__ inline v16bf ld_frag(const __bf16* tile, int row0, int lane) {
  const __bf16* p = tile + (row0 + (lane & 15)) * LDT + ((lane >> 4) * 16);
  union { u32x4 q[2]; v16bf v; } f;
  f.q[0] = *(const u32x4*)(p);
  f.q[1] = *(const u32x4*)(p + 8);
  return f.v;
}

__device__ inline v8f wmma_bf16(v16bf a, v16bf b, v8f c) {
  return __builtin_amdgcn_wmma_f32_16x16x32_bf16(
      /*neg_a=*/false, a, /*neg_b=*/false, b,
      /*c_mod=*/(short)0, c, /*reuse_a=*/false, /*reuse_b=*/false);
}

// Stage B tile (BN x BK) from f32 weight stored [OUT, K] (K fastest): rows contiguous.
__device__ inline void stage_B_kfast(__bf16* Bl, const float* __restrict__ W,
                                     long ldw, int j0, int k0, int tid) {
#pragma unroll
  for (int p = 0; p < 4; ++p) {
    int idx = tid + p * 256;           // 1024 float4-quads of 4
    int j   = idx >> 3;                // 0..127
    int kq  = (idx & 7) * 4;           // 0..28
    const float* src = W + (long)(j0 + j) * ldw + k0 + kq;
    f32x4 v = *(const f32x4*)src;
    __builtin_prefetch(src + BK, 0, 1);   // next K-step (global_prefetch_b8)
    u32x2 pk;
    pk.x = pk2bf(v.x, v.y);
    pk.y = pk2bf(v.z, v.w);
    *(u32x2*)(Bl + j * LDT + kq) = pk;
  }
}

// Stage B tile (BN x BK) from f32 weight stored [K, OUT] (OUT fastest): transpose into LDS.
__device__ inline void stage_B_ofast(__bf16* Bl, const float* __restrict__ W,
                                     long ldw, int j0, int k0, int tid) {
#pragma unroll
  for (int p = 0; p < 4; ++p) {
    int idx = tid + p * 256;
    int k   = idx >> 5;                // 0..31
    int jq  = (idx & 31) * 4;          // 0..124
    const float* src = W + (long)(k0 + k) * ldw + j0 + jq;
    f32x4 v = *(const f32x4*)src;
    __builtin_prefetch(src + (long)BK * ldw, 0, 1);  // next K-step
    Bl[(jq + 0) * LDT + k] = (__bf16)v.x;
    Bl[(jq + 1) * LDT + k] = (__bf16)v.y;
    Bl[(jq + 2) * LDT + k] = (__bf16)v.z;
    Bl[(jq + 3) * LDT + k] = (__bf16)v.w;
  }
}

// ---------------- prep: f32 -> bf16 cast of x, zero expert counters ----------------
__global__ void __launch_bounds__(256)
prep_kernel(const float* __restrict__ x, __bf16* __restrict__ xb, int* counts) {
  long i0 = ((long)blockIdx.x * 256 + threadIdx.x) * 8;
  f32x4 a = *(const f32x4*)(x + i0);
  f32x4 b = *(const f32x4*)(x + i0 + 4);
  u32x4 o;
  o.x = pk2bf(a.x, a.y); o.y = pk2bf(a.z, a.w);
  o.z = pk2bf(b.x, b.y); o.w = pk2bf(b.z, b.w);
  *(u32x4*)(xb + i0) = o;
  if (blockIdx.x == 0 && threadIdx.x < E_NUM) counts[threadIdx.x] = 0;
}

// ---------------- router: top-2 of 8 logits, sigmoid gates, bucket append ----------------
__global__ void __launch_bounds__(256)
router_kernel(const float* __restrict__ x, const float* __restrict__ rw,
              int* counts, int* tokens, float* gbuf) {
  const int wave = threadIdx.x >> 5;
  const int lane = threadIdx.x & 31;
  const int n = blockIdx.x * 8 + wave;
  const float* xr = x + (long)n * C_DIM;
  float acc[E_NUM];
#pragma unroll
  for (int e = 0; e < E_NUM; ++e) acc[e] = 0.0f;
  for (int k = lane; k < C_DIM; k += 32) {
    float xv = xr[k];
#pragma unroll
    for (int e = 0; e < E_NUM; ++e) acc[e] = fmaf(xv, rw[e * C_DIM + k], acc[e]);
  }
#pragma unroll
  for (int e = 0; e < E_NUM; ++e) {
#pragma unroll
    for (int off = 16; off > 0; off >>= 1) acc[e] += __shfl_xor(acc[e], off, 32);
  }
  if (lane == 0) {
    int i1 = 0; float v1 = acc[0];
#pragma unroll
    for (int e = 1; e < E_NUM; ++e) if (acc[e] > v1) { v1 = acc[e]; i1 = e; }
    int i2 = -1; float v2 = -3.4e38f;
#pragma unroll
    for (int e = 0; e < E_NUM; ++e) if (e != i1 && acc[e] > v2) { v2 = acc[e]; i2 = e; }
    float g1 = sigmoid_f(v1);
    float g2 = sigmoid_f(v2);
    int s1 = atomicAdd(&counts[i1], 1);
    tokens[i1 * CAP + s1] = n; gbuf[i1 * CAP + s1] = g1;
    int s2 = atomicAdd(&counts[i2], 1);
    tokens[i2 * CAP + s2] = n; gbuf[i2 * CAP + s2] = g2;
  }
}

// ---------------- shared expert: hs = silu(x Wg^T) * (x Wu^T), bf16 out ----------------
// Double-buffered LDS; A tile fetched by the Tensor Data Mover (wave 0 issues,
// TENSORcnt-waited before its barrier signal).
__global__ void __launch_bounds__(256)
shared_gateup_kernel(const __bf16* __restrict__ xb, const float* __restrict__ wg,
                     const float* __restrict__ wu, __bf16* __restrict__ hs) {
  __shared__ __align__(16) __bf16 Alds[2][BM * LDT];
  __shared__ __align__(16) __bf16 Bgl[2][BN * LDT];
  __shared__ __align__(16) __bf16 Bul[2][BN * LDT];
  const int tid = threadIdx.x, lane = tid & 31, wave = tid >> 5;
  const int wm = (wave >> 2) * 32, wn = (wave & 3) * 32;
  const int m0 = blockIdx.y * BM, n0 = blockIdx.x * BN;

  const v8f vz = {0.f, 0.f, 0.f, 0.f, 0.f, 0.f, 0.f, 0.f};
  v8f accg[2][2], accu[2][2];
#pragma unroll
  for (int i = 0; i < 2; ++i)
#pragma unroll
    for (int j = 0; j < 2; ++j) { accg[i][j] = vz; accu[i][j] = vz; }

  const __bf16* atile = xb + (long)m0 * C_DIM;

  for (int kb = 0, s = 0; kb < C_DIM; kb += BK, s ^= 1) {
    if (tid < 32) {                                   // one wave drives the TDM
      tdm_load_tile_2d(atile + kb, Alds[s], BK, BM, C_DIM);
    }
    stage_B_kfast(Bgl[s], wg, C_DIM, n0, kb, tid);
    stage_B_kfast(Bul[s], wu, C_DIM, n0, kb, tid);
    if (tid < 32) wait_tensor0();
    __syncthreads();
    v16bf a0 = ld_frag(Alds[s], wm, lane), a1 = ld_frag(Alds[s], wm + 16, lane);
    v16bf g0 = ld_frag(Bgl[s], wn, lane),  g1 = ld_frag(Bgl[s], wn + 16, lane);
    v16bf u0 = ld_frag(Bul[s], wn, lane),  u1 = ld_frag(Bul[s], wn + 16, lane);
    accg[0][0] = wmma_bf16(a0, g0, accg[0][0]);
    accg[0][1] = wmma_bf16(a0, g1, accg[0][1]);
    accg[1][0] = wmma_bf16(a1, g0, accg[1][0]);
    accg[1][1] = wmma_bf16(a1, g1, accg[1][1]);
    accu[0][0] = wmma_bf16(a0, u0, accu[0][0]);
    accu[0][1] = wmma_bf16(a0, u1, accu[0][1]);
    accu[1][0] = wmma_bf16(a1, u0, accu[1][0]);
    accu[1][1] = wmma_bf16(a1, u1, accu[1][1]);
  }
#pragma unroll
  for (int mi = 0; mi < 2; ++mi)
#pragma unroll
    for (int ni = 0; ni < 2; ++ni) {
      int rbase = m0 + wm + mi * 16 + (lane >> 4) * 8;
      int col   = n0 + wn + ni * 16 + (lane & 15);
#pragma unroll
      for (int r = 0; r < 8; ++r) {
        float h = silu_f(accg[mi][ni][r]) * accu[mi][ni][r];
        hs[(long)(rbase + r) * HS_DIM + col] = (__bf16)h;
      }
    }
}

// ---------------- shared expert down: out = hs @ Wd^T (plain f32 store) ----------------
__global__ void __launch_bounds__(256)
shared_down_kernel(const __bf16* __restrict__ hs, const float* __restrict__ wd,
                   float* __restrict__ out) {
  __shared__ __align__(16) __bf16 Alds[2][BM * LDT];
  __shared__ __align__(16) __bf16 Bl[2][BN * LDT];
  const int tid = threadIdx.x, lane = tid & 31, wave = tid >> 5;
  const int wm = (wave >> 2) * 32, wn = (wave & 3) * 32;
  const int m0 = blockIdx.y * BM, n0 = blockIdx.x * BN;

  const v8f vz = {0.f, 0.f, 0.f, 0.f, 0.f, 0.f, 0.f, 0.f};
  v8f acc[2][2];
#pragma unroll
  for (int i = 0; i < 2; ++i)
#pragma unroll
    for (int j = 0; j < 2; ++j) acc[i][j] = vz;

  const __bf16* atile = hs + (long)m0 * HS_DIM;

  for (int kb = 0, s = 0; kb < HS_DIM; kb += BK, s ^= 1) {
    if (tid < 32) {
      tdm_load_tile_2d(atile + kb, Alds[s], BK, BM, HS_DIM);
    }
    stage_B_kfast(Bl[s], wd, HS_DIM, n0, kb, tid);
    if (tid < 32) wait_tensor0();
    __syncthreads();
    v16bf a0 = ld_frag(Alds[s], wm, lane), a1 = ld_frag(Alds[s], wm + 16, lane);
    v16bf b0 = ld_frag(Bl[s], wn, lane),   b1 = ld_frag(Bl[s], wn + 16, lane);
    acc[0][0] = wmma_bf16(a0, b0, acc[0][0]);
    acc[0][1] = wmma_bf16(a0, b1, acc[0][1]);
    acc[1][0] = wmma_bf16(a1, b0, acc[1][0]);
    acc[1][1] = wmma_bf16(a1, b1, acc[1][1]);
  }
#pragma unroll
  for (int mi = 0; mi < 2; ++mi)
#pragma unroll
    for (int ni = 0; ni < 2; ++ni) {
      int rbase = m0 + wm + mi * 16 + (lane >> 4) * 8;
      int col   = n0 + wn + ni * 16 + (lane & 15);
#pragma unroll
      for (int r = 0; r < 8; ++r)
        out[(long)(rbase + r) * C_DIM + col] = acc[mi][ni][r];
    }
}

// ---------------- expert gate_up (gathered rows) + SwiGLU -> hidden bf16 ----------------
// Rows are token-gathered (not a dense tile), so use per-lane async LDS loads.
__global__ void __launch_bounds__(256)
expert_gateup_kernel(const __bf16* __restrict__ xb, const float* __restrict__ gup,
                     const int* __restrict__ counts, const int* __restrict__ tokens,
                     __bf16* __restrict__ hidden) {
  const int e = blockIdx.z;
  const int cnt = counts[e];
  const int m0 = blockIdx.y * BM;
  if (m0 >= cnt) return;
  const int n0 = blockIdx.x * BN;
  const float* W = gup + (long)e * C_DIM * (2 * H_DIM);
  const int* tk = tokens + e * CAP;

  __shared__ __align__(16) __bf16 Alds[2][BM * LDT];
  __shared__ __align__(16) __bf16 Bgl[2][BN * LDT];
  __shared__ __align__(16) __bf16 Bul[2][BN * LDT];
  const int tid = threadIdx.x, lane = tid & 31, wave = tid >> 5;
  const int wm = (wave >> 2) * 32, wn = (wave & 3) * 32;

  const v8f vz = {0.f, 0.f, 0.f, 0.f, 0.f, 0.f, 0.f, 0.f};
  v8f accg[2][2], accu[2][2];
#pragma unroll
  for (int i = 0; i < 2; ++i)
#pragma unroll
    for (int j = 0; j < 2; ++j) { accg[i][j] = vz; accu[i][j] = vz; }

  const int ar = tid >> 2, akq = (tid & 3) * 8;
  const int aoff = ar * LDT + akq;
  const long arow = (m0 + ar < cnt) ? (long)tk[m0 + ar] : 0L;
  const __bf16* asrc = xb + arow * C_DIM + akq;

  for (int kb = 0, s = 0; kb < C_DIM; kb += BK, s ^= 1) {
    async_copy16(Alds[s] + aoff, asrc + kb);                   // gathered token row
    stage_B_ofast(Bgl[s], W, 2 * H_DIM, n0, kb, tid);          // gate cols
    stage_B_ofast(Bul[s], W, 2 * H_DIM, H_DIM + n0, kb, tid);  // up cols
    wait_async0();
    __syncthreads();
    v16bf a0 = ld_frag(Alds[s], wm, lane), a1 = ld_frag(Alds[s], wm + 16, lane);
    v16bf g0 = ld_frag(Bgl[s], wn, lane),  g1 = ld_frag(Bgl[s], wn + 16, lane);
    v16bf u0 = ld_frag(Bul[s], wn, lane),  u1 = ld_frag(Bul[s], wn + 16, lane);
    accg[0][0] = wmma_bf16(a0, g0, accg[0][0]);
    accg[0][1] = wmma_bf16(a0, g1, accg[0][1]);
    accg[1][0] = wmma_bf16(a1, g0, accg[1][0]);
    accg[1][1] = wmma_bf16(a1, g1, accg[1][1]);
    accu[0][0] = wmma_bf16(a0, u0, accu[0][0]);
    accu[0][1] = wmma_bf16(a0, u1, accu[0][1]);
    accu[1][0] = wmma_bf16(a1, u0, accu[1][0]);
    accu[1][1] = wmma_bf16(a1, u1, accu[1][1]);
  }
#pragma unroll
  for (int mi = 0; mi < 2; ++mi)
#pragma unroll
    for (int ni = 0; ni < 2; ++ni) {
      int sbase = m0 + wm + mi * 16 + (lane >> 4) * 8;
      int col   = n0 + wn + ni * 16 + (lane & 15);
#pragma unroll
      for (int r = 0; r < 8; ++r) {
        int slot = sbase + r;
        if (slot < cnt) {
          float h = silu_f(accg[mi][ni][r]) * accu[mi][ni][r];
          hidden[((long)e * CAP + slot) * H_DIM + col] = (__bf16)h;
        }
      }
    }
}

// ---------------- expert down + gated atomic scatter into out ----------------
__global__ void __launch_bounds__(256)
expert_down_kernel(const __bf16* __restrict__ hidden, const float* __restrict__ dw,
                   const int* __restrict__ counts, const int* __restrict__ tokens,
                   const float* __restrict__ gbuf, float* __restrict__ out) {
  const int e = blockIdx.z;
  const int cnt = counts[e];
  const int m0 = blockIdx.y * BM;
  if (m0 >= cnt) return;
  const int n0 = blockIdx.x * BN;
  const float* W = dw + (long)e * H_DIM * C_DIM;
  const __bf16* A = hidden + (long)e * CAP * H_DIM;

  __shared__ __align__(16) __bf16 Alds[2][BM * LDT];
  __shared__ __align__(16) __bf16 Bl[2][BN * LDT];
  const int tid = threadIdx.x, lane = tid & 31, wave = tid >> 5;
  const int wm = (wave >> 2) * 32, wn = (wave & 3) * 32;

  const v8f vz = {0.f, 0.f, 0.f, 0.f, 0.f, 0.f, 0.f, 0.f};
  v8f acc[2][2];
#pragma unroll
  for (int i = 0; i < 2; ++i)
#pragma unroll
    for (int j = 0; j < 2; ++j) acc[i][j] = vz;

  const int ar = tid >> 2, akq = (tid & 3) * 8;
  const int aoff = ar * LDT + akq;
  const __bf16* asrc = A + (long)(m0 + ar) * H_DIM + akq;

  for (int kb = 0, s = 0; kb < H_DIM; kb += BK, s ^= 1) {
    async_copy16(Alds[s] + aoff, asrc + kb);
    stage_B_ofast(Bl[s], W, C_DIM, n0, kb, tid);
    wait_async0();
    __syncthreads();
    v16bf a0 = ld_frag(Alds[s], wm, lane), a1 = ld_frag(Alds[s], wm + 16, lane);
    v16bf b0 = ld_frag(Bl[s], wn, lane),   b1 = ld_frag(Bl[s], wn + 16, lane);
    acc[0][0] = wmma_bf16(a0, b0, acc[0][0]);
    acc[0][1] = wmma_bf16(a0, b1, acc[0][1]);
    acc[1][0] = wmma_bf16(a1, b0, acc[1][0]);
    acc[1][1] = wmma_bf16(a1, b1, acc[1][1]);
  }
#pragma unroll
  for (int mi = 0; mi < 2; ++mi)
#pragma unroll
    for (int ni = 0; ni < 2; ++ni) {
      int sbase = m0 + wm + mi * 16 + (lane >> 4) * 8;
      int col   = n0 + wn + ni * 16 + (lane & 15);
#pragma unroll
      for (int r = 0; r < 8; ++r) {
        int slot = sbase + r;
        if (slot < cnt) {
          int tok = tokens[e * CAP + slot];
          float g = gbuf[e * CAP + slot];
          atomicAdd(&out[(long)tok * C_DIM + col], g * acc[mi][ni][r]);
        }
      }
    }
}

// ---------------- launcher ----------------
extern "C" void kernel_launch(void* const* d_in, const int* in_sizes, int n_in,
                              void* d_out, int out_size, void* d_ws, size_t ws_size,
                              hipStream_t stream) {
  (void)in_sizes; (void)n_in; (void)out_size; (void)ws_size;
  const float* x   = (const float*)d_in[0];
  const float* rw  = (const float*)d_in[1];
  const float* gup = (const float*)d_in[2];
  const float* dwn = (const float*)d_in[3];
  const float* sg  = (const float*)d_in[4];
  const float* su  = (const float*)d_in[5];
  const float* sd  = (const float*)d_in[6];
  float* out = (float*)d_out;

  char* ws = (char*)d_ws;
  __bf16* xb     = (__bf16*)ws;  ws += (size_t)N_TOK * C_DIM * 2;
  __bf16* hs     = (__bf16*)ws;  ws += (size_t)N_TOK * HS_DIM * 2;
  __bf16* hidden = (__bf16*)ws;  ws += (size_t)E_NUM * CAP * H_DIM * 2;
  int*    tokens = (int*)ws;     ws += (size_t)E_NUM * CAP * 4;
  float*  gbuf   = (float*)ws;   ws += (size_t)E_NUM * CAP * 4;
  int*    counts = (int*)ws;

  // 1) cast x -> bf16, zero counters
  prep_kernel<<<(N_TOK * C_DIM) / (256 * 8), 256, 0, stream>>>(x, xb, counts);
  // 2) router: build per-expert token buckets
  router_kernel<<<N_TOK / 8, 256, 0, stream>>>(x, rw, counts, tokens, gbuf);
  // 3) shared expert SwiGLU hidden (TDM A-tile staging)
  shared_gateup_kernel<<<dim3(HS_DIM / BN, N_TOK / BM), 256, 0, stream>>>(xb, sg, su, hs);
  // 4) shared expert down (fully initializes d_out; TDM A-tile staging)
  shared_down_kernel<<<dim3(C_DIM / BN, N_TOK / BM), 256, 0, stream>>>(hs, sd, out);
  // 5) gathered expert SwiGLU hidden (async-LDS gathered A)
  expert_gateup_kernel<<<dim3(H_DIM / BN, N_TOK / BM, E_NUM), 256, 0, stream>>>(
      xb, gup, counts, tokens, hidden);
  // 6) expert down + gated atomic scatter
  expert_down_kernel<<<dim3(C_DIM / BN, N_TOK / BM, E_NUM), 256, 0, stream>>>(
      hidden, dwn, counts, tokens, gbuf, out);
}
